// SWAttention_67989332295880
// MI455X (gfx1250) — matrix operand order repaired and checked by reading, not compiled, not run on hardware
//
#include <hip/hip_runtime.h>
#include <hip/hip_bf16.h>
#include <math.h>

// Problem constants (reference: B=2, S=1024, HID=2048, H=32, KVH=8, D=64, MAXB=1024)
#define BATCH 2
#define SEQ   1024
#define HID   2048
#define NH    32
#define NKVH  8
#define HD    64
#define MAXB  1024

typedef __attribute__((ext_vector_type(16))) __bf16      v16bf;
typedef __attribute__((ext_vector_type(8)))  float       v8f;
typedef __attribute__((ext_vector_type(4)))  unsigned int u32x4;
typedef __attribute__((ext_vector_type(4)))  float       f32x4;

struct U32x8 { u32x4 a, b; };

__device__ __forceinline__ unsigned short f2bf(float f) {
  return __builtin_bit_cast(unsigned short, (__bf16)f);   // native RNE convert on gfx1250
}

// Async 16-byte global -> LDS copy (ASYNCcnt-tracked).  LDS offset is the low
// 32 bits of the generic pointer (LDS aperture maps offset into addr[31:0]).
__device__ __forceinline__ void async_copy16(const void* gptr, void* lptr) {
  unsigned int lds = (unsigned int)(size_t)lptr;
  asm volatile("global_load_async_to_lds_b128 %0, %1, off"
               :: "v"(lds), "v"(gptr) : "memory");
}
__device__ __forceinline__ void wait_async() {
  asm volatile("s_wait_asynccnt 0x0" ::: "memory");
}

// A-operand fragment (16x32 bf16, M x K): lane m = lane&15,
// K elems {kb..kb+7, kb+16..kb+23} with kb = 8*(lane>>4).  Tile M-major, K-contiguous.
__device__ __forceinline__ v16bf ld_fragA(const unsigned short* tile, int stride, int lane) {
  int m  = lane & 15;
  int kb = (lane >> 4) << 3;
  const unsigned short* p = tile + m * stride + kb;
  U32x8 u;
  u.a = *(const u32x4*)(p);
  u.b = *(const u32x4*)(p + 16);
  return __builtin_bit_cast(v16bf, u);
}

// B-operand fragment (32x16 bf16, K x N): lane n = lane&15,
// K elems {kb..kb+7, kb+16..kb+23} with kb = 8*(lane>>4).  Tile N-major, K-contiguous.
__device__ __forceinline__ v16bf ld_fragB(const unsigned short* tile, int stride, int lane) {
  int n  = lane & 15;
  int kb = (lane >> 4) << 3;
  const unsigned short* p = tile + n * stride + kb;
  U32x8 u;
  u.a = *(const u32x4*)(p);
  u.b = *(const u32x4*)(p + 16);
  return __builtin_bit_cast(v16bf, u);
}

__device__ __forceinline__ v8f wmma_bf16(v16bf a, v16bf b, v8f c) {
  return __builtin_amdgcn_wmma_f32_16x16x32_bf16(false, a, false, b, (short)0, c, false, false);
}

// ---------------------------------------------------------------------------
// Bulk f32 -> bf16 conversion, 8 elements/thread (n % 2048 == 0 for all uses).
// ---------------------------------------------------------------------------
__global__ __launch_bounds__(256) void conv_bf16_kernel(const float* __restrict__ in,
                                                        unsigned short* __restrict__ out,
                                                        int n) {
  int i = (blockIdx.x * 256 + threadIdx.x) * 8;
  if (i >= n) return;
  f32x4 a = *(const f32x4*)(in + i);
  f32x4 b = *(const f32x4*)(in + i + 4);
  u32x4 o;
  o.x = (unsigned)f2bf(a.x) | ((unsigned)f2bf(a.y) << 16);
  o.y = (unsigned)f2bf(a.z) | ((unsigned)f2bf(a.w) << 16);
  o.z = (unsigned)f2bf(b.x) | ((unsigned)f2bf(b.y) << 16);
  o.w = (unsigned)f2bf(b.z) | ((unsigned)f2bf(b.w) << 16);
  *(u32x4*)(out + i) = o;
}

// ---------------------------------------------------------------------------
// Tiled GEMM: C[M,N] = A[M,K] * B[K,N], A/B bf16 in global, f32 out.
// Block tile 128x128, K-step 32.  8 waves in a 4x2 grid; wave tile 32x64 =
// 2x4 WMMA accumulators (8 WMMAs per K-step).  Double-buffered LDS; A tile
// streamed with global_load_async_to_lds_b128, B tile transposed via VGPRs.
// Requires M%128==0, N%128==0, K%32==0.
// ---------------------------------------------------------------------------
__global__ __launch_bounds__(256) void gemm_bf16_kernel(const unsigned short* __restrict__ A,
                                                        const unsigned short* __restrict__ B,
                                                        float* __restrict__ C,
                                                        int M, int N, int K) {
  __shared__ __align__(16) unsigned short As[2][128 * 40];  // 128 x 32, stride 40
  __shared__ __align__(16) unsigned short Bt[2][128 * 40];  // N-major: 128 x 32

  const int tid  = threadIdx.x;
  const int lane = tid & 31;
  const int wave = tid >> 5;
  const int wm   = wave >> 1;   // 0..3
  const int wn   = wave & 1;    // 0..1
  const int m0   = blockIdx.y * 128;
  const int n0   = blockIdx.x * 128;
  const int T    = K >> 5;

  v8f acc[2][4] = {};

  // ---- stage tile 0 ----
  {
#pragma unroll
    for (int i = 0; i < 2; ++i) {             // A: 512 16B chunks, 2/thread
      int c = tid + i * 256;
      int row = c >> 2, cq = (c & 3) << 3;
      async_copy16(A + (size_t)(m0 + row) * K + cq, &As[0][row * 40 + cq]);
    }
    const unsigned int* Bg32 = (const unsigned int*)B;
#pragma unroll
    for (int i = 0; i < 8; ++i) {             // B: 2048 dwords, 8/thread
      int idx = tid + i * 256;
      int kr = idx >> 6, nd2 = idx & 63;
      unsigned int v = Bg32[((size_t)kr * N + n0) / 2 + nd2];
      Bt[0][(nd2 * 2) * 40 + kr]     = (unsigned short)(v & 0xFFFFu);
      Bt[0][(nd2 * 2 + 1) * 40 + kr] = (unsigned short)(v >> 16);
    }
    wait_async();
    __syncthreads();
  }

  for (int kt = 0; kt < T; ++kt) {
    const int cur = kt & 1, nxt = cur ^ 1;
    const bool have_next = (kt + 1) < T;
    unsigned int breg[8];

    if (have_next) {
      const int k0n = (kt + 1) * 32;
#pragma unroll
      for (int i = 0; i < 2; ++i) {
        int c = tid + i * 256;
        int row = c >> 2, cq = (c & 3) << 3;
        async_copy16(A + (size_t)(m0 + row) * K + k0n + cq, &As[nxt][row * 40 + cq]);
      }
      const unsigned int* Bg32 = (const unsigned int*)B;
#pragma unroll
      for (int i = 0; i < 8; ++i) {
        int idx = tid + i * 256;
        int kr = idx >> 6, nd2 = idx & 63;
        breg[i] = Bg32[((size_t)(k0n + kr) * N + n0) / 2 + nd2];
      }
    }

    // ---- compute on current buffers (overlaps the loads above) ----
#pragma unroll
    for (int i = 0; i < 2; ++i) {
      v16bf a = ld_fragA(&As[cur][(wm * 32 + i * 16) * 40], 40, lane);
#pragma unroll
      for (int j = 0; j < 4; ++j) {
        v16bf b = ld_fragB(&Bt[cur][(wn * 64 + j * 16) * 40], 40, lane);
        acc[i][j] = wmma_bf16(a, b, acc[i][j]);
      }
    }

    if (have_next) {
#pragma unroll
      for (int i = 0; i < 8; ++i) {
        int idx = tid + i * 256;
        int kr = idx >> 6, nd2 = idx & 63;
        Bt[nxt][(nd2 * 2) * 40 + kr]     = (unsigned short)(breg[i] & 0xFFFFu);
        Bt[nxt][(nd2 * 2 + 1) * 40 + kr] = (unsigned short)(breg[i] >> 16);
      }
    }
    wait_async();
    __syncthreads();
  }

  const int nl = lane & 15, hi = lane >> 4;
#pragma unroll
  for (int i = 0; i < 2; ++i)
#pragma unroll
    for (int j = 0; j < 4; ++j)
#pragma unroll
      for (int r = 0; r < 8; ++r) {
        int m = m0 + wm * 32 + i * 16 + r + 8 * hi;
        int n = n0 + wn * 64 + j * 16 + nl;
        C[(size_t)m * N + n] = acc[i][j][r];
      }
}

// ---------------------------------------------------------------------------
// RoPE (optional) + f32 -> bf16 + layout [B,S,h,D] -> [B,h,S,D]
// ---------------------------------------------------------------------------
__global__ __launch_bounds__(256) void rope_conv_kernel(const float* __restrict__ in,
                                                        unsigned short* __restrict__ out,
                                                        int heads, int apply_rope) {
  int idx = blockIdx.x * blockDim.x + threadIdx.x;
  int total = BATCH * SEQ * heads * HD;
  if (idx >= total) return;
  int d = idx & 63;
  int h = (idx >> 6) % heads;
  int s = (idx / (64 * heads)) % SEQ;
  int b = idx / (64 * heads * SEQ);
  float x = in[idx];
  float val = x;
  if (apply_rope) {
    int j = d & 31;
    float inv = __powf(10000.0f, -(float)(2 * j) / 64.0f);
    float ang = (float)s * inv;
    float c, sn;
    __sincosf(ang, &sn, &c);
    if (d < 32) { float x2 = in[idx + 32]; val = x * c - x2 * sn; }
    else        { float x1 = in[idx - 32]; val = x * c + x1 * sn; }
  }
  out[(((size_t)b * heads + h) * SEQ + s) * HD + d] = f2bf(val);
}

// ---------------------------------------------------------------------------
// Two-pass flash attention with post-softmax relu-offset.
// Grid (SEQ/128, NH, BATCH), 256 threads = 8 waves; wave w owns 16 q rows.
// Q/K tiles streamed via global_load_async_to_lds_b128 (bf16, no transpose);
// V transposed through VGPRs.  Emits bf16 attn_out for the O-projection.
// ---------------------------------------------------------------------------
__global__ __launch_bounds__(256) void flash_attn_kernel(const unsigned short* __restrict__ Qb,
                                                         const unsigned short* __restrict__ Kb,
                                                         const unsigned short* __restrict__ Vb,
                                                         const float* __restrict__ bias_diag,
                                                         const float* __restrict__ soff,
                                                         unsigned short* __restrict__ Oout) {
  __shared__ __align__(16) unsigned short Qs[128 * 72];
  __shared__ __align__(16) unsigned short Ks[32 * 72];
  __shared__ __align__(16) unsigned short Vt[64 * 40];
  __shared__ __align__(16) unsigned short Ps[8][16 * 40];
  __shared__ float Bs[MAXB];

  const int tid  = threadIdx.x;
  const int lane = tid & 31;
  const int wave = tid >> 5;
  const int qblk = blockIdx.x;
  const int h    = blockIdx.y;
  const int b    = blockIdx.z;
  const int kvh  = h >> 2;                 // H/KVH = 4
  const int q0   = qblk * 128;

  const unsigned short* Qg = Qb + (((size_t)b * NH   + h)   * SEQ + q0) * HD;
  const unsigned short* Kg = Kb +  ((size_t)b * NKVH + kvh) * SEQ * HD;
  const unsigned short* Vg = Vb +  ((size_t)b * NKVH + kvh) * SEQ * HD;

  for (int i = tid; i < MAXB; i += 256) Bs[i] = bias_diag[(size_t)h * MAXB + i];

  // Q tile resident: 1024 16B chunks, 4/thread, async to LDS
#pragma unroll
  for (int i = 0; i < 4; ++i) {
    int c = tid + i * 256;
    int row = c >> 3, cq = (c & 7) << 3;
    async_copy16(Qg + (size_t)row * HD + cq, &Qs[row * 72 + cq]);
  }
  wait_async();
  __syncthreads();

  const int nl = lane & 15, hig = lane >> 4;
  const int mBase = q0 + wave * 16;
  const float scaling = 0.125f;            // 64^-0.5
  const int kTiles = (q0 + 128) >> 5;

  float Mrow[8], Zrow[8];
#pragma unroll
  for (int r = 0; r < 8; ++r) { Mrow[r] = -INFINITY; Zrow[r] = 0.0f; }

  // ---------------- Pass 1: row max + Z ----------------
  for (int kt = 0; kt < kTiles; ++kt) {
    const int kb = kt * 32;
    {
      int row = tid >> 3, cq = (tid & 7) << 3;     // 256 chunks, 1/thread
      async_copy16(Kg + (size_t)(kb + row) * HD + cq, &Ks[row * 72 + cq]);
    }
    wait_async();
    __syncthreads();

    v16bf a0 = ld_fragA(&Qs[(wave * 16) * 72 + 0], 72, lane);
    v16bf a1 = ld_fragA(&Qs[(wave * 16) * 72 + 32], 72, lane);

    float sc[2][8];
#pragma unroll
    for (int sub = 0; sub < 2; ++sub) {
      v8f c = {};
      v16bf b0 = ld_fragB(&Ks[(sub * 16) * 72 + 0], 72, lane);
      v16bf b1 = ld_fragB(&Ks[(sub * 16) * 72 + 32], 72, lane);
      c = wmma_bf16(a0, b0, c);
      c = wmma_bf16(a1, b1, c);
#pragma unroll
      for (int r = 0; r < 8; ++r) {
        int gq = mBase + r + 8 * hig;
        int gk = kb + sub * 16 + nl;
        sc[sub][r] = (gk > gq) ? -INFINITY : (c[r] * scaling + Bs[gq - gk]);
      }
    }
#pragma unroll
    for (int r = 0; r < 8; ++r) {
      float mx = fmaxf(sc[0][r], sc[1][r]);
#pragma unroll
      for (int off = 1; off < 16; off <<= 1) mx = fmaxf(mx, __shfl_xor(mx, off, 32));
      float Mo = Mrow[r];
      float Mn = fmaxf(Mo, mx);
      float e0 = (sc[0][r] == -INFINITY) ? 0.0f : __expf(sc[0][r] - Mn);
      float e1 = (sc[1][r] == -INFINITY) ? 0.0f : __expf(sc[1][r] - Mn);
      float p  = e0 + e1;
#pragma unroll
      for (int off = 1; off < 16; off <<= 1) p += __shfl_xor(p, off, 32);
      float corr = (Mo == -INFINITY) ? 0.0f : __expf(Mo - Mn);
      Zrow[r] = Zrow[r] * corr + p;
      Mrow[r] = Mn;
    }
    __syncthreads();
  }

  // ---------------- Pass 2: weights & P*V ----------------
  const float offs = fabsf(soff[h] + 1.0f);
  v8f o[4] = {};

  for (int kt = 0; kt < kTiles; ++kt) {
    const int kb = kt * 32;
    {
      int row = tid >> 3, cq = (tid & 7) << 3;
      async_copy16(Kg + (size_t)(kb + row) * HD + cq, &Ks[row * 72 + cq]);
    }
    // V tile transposed: Vt[d][k] (1024 dwords, 4/thread)
    const unsigned int* Vg32 = (const unsigned int*)(Vg + (size_t)kb * HD);
#pragma unroll
    for (int i = 0; i < 4; ++i) {
      int idx = tid + i * 256;
      int row = idx >> 5, d2 = (idx & 31) * 2;
      unsigned int v = Vg32[row * 32 + (idx & 31)];
      Vt[d2 * 40 + row]       = (unsigned short)(v & 0xFFFFu);
      Vt[(d2 + 1) * 40 + row] = (unsigned short)(v >> 16);
    }
    wait_async();
    __syncthreads();

    v16bf a0 = ld_fragA(&Qs[(wave * 16) * 72 + 0], 72, lane);
    v16bf a1 = ld_fragA(&Qs[(wave * 16) * 72 + 32], 72, lane);
    unsigned short* Pw = &Ps[wave][0];

#pragma unroll
    for (int sub = 0; sub < 2; ++sub) {
      v8f c = {};
      v16bf b0 = ld_fragB(&Ks[(sub * 16) * 72 + 0], 72, lane);
      v16bf b1 = ld_fragB(&Ks[(sub * 16) * 72 + 32], 72, lane);
      c = wmma_bf16(a0, b0, c);
      c = wmma_bf16(a1, b1, c);
#pragma unroll
      for (int r = 0; r < 8; ++r) {
        int m  = r + 8 * hig;
        int gq = mBase + m;
        int gk = kb + sub * 16 + nl;
        float w = 0.0f;
        if (gk <= gq) {
          float s  = c[r] * scaling + Bs[gq - gk];
          float pn = __expf(s - Mrow[r]) / Zrow[r];
          w = fmaxf(pn - offs / (float)(gq + 1), 0.0f);
        }
        Pw[m * 40 + sub * 16 + nl] = f2bf(w);
      }
    }
    // Same-wave LDS ops are in-order -> safe to re-fragment P immediately.
    v16bf ap = ld_fragA(Pw, 40, lane);
#pragma unroll
    for (int sub = 0; sub < 4; ++sub) {
      v16bf bv = ld_fragB(&Vt[(sub * 16) * 40], 40, lane);
      o[sub] = wmma_bf16(ap, bv, o[sub]);
    }
    __syncthreads();
  }

  // attn_out as bf16 [B, S, H*D] (A-operand of the O-projection GEMM)
#pragma unroll
  for (int sub = 0; sub < 4; ++sub)
#pragma unroll
    for (int r = 0; r < 8; ++r) {
      int gq = mBase + r + 8 * hig;
      int d  = sub * 16 + nl;
      Oout[((size_t)b * SEQ + gq) * (NH * HD) + h * HD + d] = f2bf(o[sub][r]);
    }
}

// ---------------------------------------------------------------------------
extern "C" void kernel_launch(void* const* d_in, const int* in_sizes, int n_in,
                              void* d_out, int out_size, void* d_ws, size_t ws_size,
                              hipStream_t stream) {
  (void)in_sizes; (void)n_in; (void)out_size; (void)ws_size;
  const float* hidden = (const float*)d_in[0];
  /* d_in[1] = attention_mask (causal; handled analytically) */
  const float* Wq   = (const float*)d_in[2];
  const float* Wk   = (const float*)d_in[3];
  const float* Wv   = (const float*)d_in[4];
  const float* Wo   = (const float*)d_in[5];
  const float* bias = (const float*)d_in[6];
  const float* soff = (const float*)d_in[7];
  float* out = (float*)d_out;

  char* ws = (char*)d_ws;
  unsigned short* hbf   = (unsigned short*)(ws);             //  8 MB  [2048,2048]
  unsigned short* wqbf  = (unsigned short*)(ws +  8388608);  //  8 MB
  unsigned short* wkbf  = (unsigned short*)(ws + 16777216);  //  2 MB
  unsigned short* wvbf  = (unsigned short*)(ws + 18874368);  //  2 MB
  unsigned short* wobf  = (unsigned short*)(ws + 20971520);  //  8 MB
  float*          qf    = (float*)(ws + 29360128);           // 16 MB
  float*          kf    = (float*)(ws + 46137344);           //  4 MB
  float*          vf    = (float*)(ws + 50331648);           //  4 MB
  unsigned short* qbf   = (unsigned short*)(ws + 54525952);  //  8 MB  [B,H,S,D]
  unsigned short* kbf   = (unsigned short*)(ws + 62914560);  //  2 MB  [B,KVH,S,D]
  unsigned short* vbf   = (unsigned short*)(ws + 65011712);  //  2 MB  [B,KVH,S,D]
  unsigned short* attnb = (unsigned short*)(ws + 67108864);  //  8 MB  [B,S,H*D]

  const int M = BATCH * SEQ;   // 2048

  // One-time f32 -> bf16 conversions (activations + all weights)
  conv_bf16_kernel<<<(M * HID) / 2048, 256, 0, stream>>>(hidden, hbf, M * HID);
  conv_bf16_kernel<<<(HID * NH * HD) / 2048, 256, 0, stream>>>(Wq, wqbf, HID * NH * HD);
  conv_bf16_kernel<<<(HID * NKVH * HD) / 2048, 256, 0, stream>>>(Wk, wkbf, HID * NKVH * HD);
  conv_bf16_kernel<<<(HID * NKVH * HD) / 2048, 256, 0, stream>>>(Wv, wvbf, HID * NKVH * HD);
  conv_bf16_kernel<<<(NH * HD * HID) / 2048, 256, 0, stream>>>(Wo, wobf, NH * HD * HID);

  // Projections (bf16 x bf16 -> f32)
  gemm_bf16_kernel<<<dim3((NH * HD) / 128, M / 128), 256, 0, stream>>>(hbf, wqbf, qf, M, NH * HD, HID);
  gemm_bf16_kernel<<<dim3((NKVH * HD) / 128, M / 128), 256, 0, stream>>>(hbf, wkbf, kf, M, NKVH * HD, HID);
  gemm_bf16_kernel<<<dim3((NKVH * HD) / 128, M / 128), 256, 0, stream>>>(hbf, wvbf, vf, M, NKVH * HD, HID);

  // RoPE + bf16 convert + head-major relayout
  {
    int totq = BATCH * SEQ * NH * HD;
    int totk = BATCH * SEQ * NKVH * HD;
    rope_conv_kernel<<<totq / 256, 256, 0, stream>>>(qf, qbf, NH, 1);
    rope_conv_kernel<<<totk / 256, 256, 0, stream>>>(kf, kbf, NKVH, 1);
    rope_conv_kernel<<<totk / 256, 256, 0, stream>>>(vf, vbf, NKVH, 0);
  }

  // Attention (bf16 attn_out)
  flash_attn_kernel<<<dim3(SEQ / 128, NH, BATCH), 256, 0, stream>>>(qbf, kbf, vbf, bias, soff, attnb);

  // Output projection
  gemm_bf16_kernel<<<dim3(HID / 128, M / 128), 256, 0, stream>>>(attnb, wobf, out, M, HID, HID);
}
